// AOD_Deep2_Net_77524159692831
// MI455X (gfx1250) — matrix operand order repaired and checked by reading, not compiled, not run on hardware
//
#include <hip/hip_runtime.h>

typedef __attribute__((ext_vector_type(16))) _Float16 v16h;
typedef __attribute__((ext_vector_type(8)))  _Float16 v8h;
typedef __attribute__((ext_vector_type(8)))  float    v8f;

#define NB 8
#define HH 512
#define WW 512
#define HW (HH*WW)
#define ROWS 4
#define TILE_X 16
#define LAYER_STRIDE ((size_t)NB*3*HW)   // floats per stored intermediate x_i

// One conv layer, cout==3, 'same' padding, fused bias+ReLU.
// src is interpreted as consecutive stored layers: [layer][n][3][H][W];
// concat channel c -> layer c/3, channel c%3 (all concats in this net are
// consecutive layer ranges, so a base pointer is enough).
// FINAL: compute relu(v*x - v + 1) against xin and write to dst (d_out).
template<int K, int CIN, bool FINAL>
__global__ __launch_bounds__(128)
void conv_layer(const float* __restrict__ src, const float* __restrict__ wgt,
                const float* __restrict__ bias, float* __restrict__ dst,
                const float* __restrict__ xin)
{
    constexpr int PADK = (K - 1) / 2;
    constexpr int TH   = ROWS + K - 1;
    constexpr int TW   = TILE_X + K - 1;
    constexpr int NCH  = (CIN + 31) / 32;     // K-chunks of 32
    constexpr int CINP = NCH * 32;            // padded channel count
    constexpr int SLOT = CINP * 2 + 16;       // bytes per (row,col) slot (16B-aligned, bank-skewed)
    constexpr int NT   = K * K;               // taps
    constexpr int BSLOT = 48;                 // bytes per lane in packed B (32 data + 16 pad)
    constexpr int BBYTES = NT * NCH * 32 * BSLOT;

    __shared__ __align__(16) unsigned char lds_in[TH * TW * SLOT];
    __shared__ __align__(16) unsigned char lds_b[BBYTES];

    const int tid  = threadIdx.x;
    const int lane = tid & 31;
    const int wv   = tid >> 5;        // wave 0..3 -> output row within tile
    const int h    = lane >> 4;       // lane half (K sub-range / M sub-range)
    const int mcol = lane & 15;       // A: pixel m; B/C: column n

    const int bx = blockIdx.x;
    const int x0 = (bx % (WW / TILE_X)) * TILE_X;
    const int y0 = ((bx / (WW / TILE_X)) % (HH / ROWS)) * ROWS;
    const int bn =  bx / ((WW / TILE_X) * (HH / ROWS));

    // ---- B fragments: zero-fill packed region, then scatter the real weights ----
    {
        float4 z = make_float4(0.f, 0.f, 0.f, 0.f);
        for (int idx = tid; idx < BBYTES / 16; idx += 128)
            ((float4*)lds_b)[idx] = z;
    }
    __syncthreads();
    // wgt layout (3, CIN, K, K) row-major: flat idx = (n*CIN + c)*NT + t
    for (int idx = tid; idx < 3 * CIN * NT; idx += 128) {
        int t  = idx % NT;
        int nc = idx / NT;
        int c  = nc % CIN;
        int n  = nc / CIN;
        float w  = wgt[idx];
        int kc   = c >> 5;
        int rem  = c & 31;
        int bh   = (rem >> 3) & 1;           // lane half holding this K row
        int e    = (rem & 7) | ((rem >> 4) << 3);           // element within lane
        int blan = bh * 16 + n;        // column n lives in lane (bh*16 + n)
        *(_Float16*)(lds_b + ((t * NCH + kc) * 32 + blan) * BSLOT + e * 2) = (_Float16)w;
    }

    // ---- stage input tile (f32 global -> f16 LDS in plain channel order) ----
    for (int idx = tid; idx < CIN * TH * TW; idx += 128) {
        int col = idx % TW;
        int rc  = idx / TW;
        int row = rc % TH;
        int c   = rc / TH;
        int gy = y0 - PADK + row;
        int gx = x0 - PADK + col;
        float v = 0.f;
        if (gy >= 0 && gy < HH && gx >= 0 && gx < WW) {
            int off = (((c / 3) * NB + bn) * 3 + (c % 3)) * HW + gy * WW + gx;
            v = src[off];
        }
        *(_Float16*)(lds_in + (row * TW + col) * SLOT + c * 2) = (_Float16)v;
    }
    // zero the padded channels so A fragments need no masking
    for (int idx = tid; idx < (CINP - CIN) * TH * TW; idx += 128) {
        int col = idx % TW;
        int rc  = idx / TW;
        int row = rc % TH;
        int c   = CIN + rc / TH;
        *(_Float16*)(lds_in + (row * TW + col) * SLOT + c * 2) = (_Float16)0.f;
    }
    __syncthreads();

    // ---- per-tap implicit GEMM: M=16 pixels, N=16 (3 used), K=32 channels ----
    v8f acc = {};
    #pragma unroll
    for (int dy = 0; dy < K; ++dy) {
        #pragma unroll
        for (int dx = 0; dx < K; ++dx) {
            const int t = dy * K + dx;
            const unsigned char* arow =
                lds_in + ((wv + dy) * TW + (mcol + dx)) * SLOT;
            #pragma unroll
            for (int kc = 0; kc < NCH; ++kc) {
                // A fragment: two ds_read_b128 per the 16-bit A 16x32 layout
                const v8h* ap = (const v8h*)(arow + kc * 64 + h * 16);
                v8h alo = ap[0];
                v8h ahi = ap[2];   // +32 bytes
                v16h a = __builtin_shufflevector(alo, ahi,
                    0,1,2,3,4,5,6,7,8,9,10,11,12,13,14,15);

                // B fragment: prepacked per-lane 32B -> two ds_read_b128
                const v8h* bp =
                    (const v8h*)(lds_b + ((t * NCH + kc) * 32 + lane) * BSLOT);
                v8h blo = bp[0];
                v8h bhi = bp[1];
                v16h b = __builtin_shufflevector(blo, bhi,
                    0,1,2,3,4,5,6,7,8,9,10,11,12,13,14,15);

                acc = __builtin_amdgcn_wmma_f32_16x16x32_f16(
                    false, a, false, b, (short)0, acc, false, false);
            }
        }
    }

    // ---- epilogue: bias + ReLU (+ fused dehaze for layer 20), b128 stores ----
    // C/D layout: lane holds column n=mcol; VGPR r -> row M=r (lanes<16) / r+8
    if (mcol < 3) {
        const float bv = bias[mcol];
        const int y = y0 + wv;
        const size_t outBase =
            (size_t)(bn * 3 + mcol) * HW + (size_t)y * WW + x0 + h * 8;
        float o[8];
        #pragma unroll
        for (int r = 0; r < 8; ++r) {
            float v = acc[r] + bv;
            o[r] = v > 0.f ? v : 0.f;
        }
        if (FINAL) {
            const float4* xp = (const float4*)(xin + outBase);
            float4 xa = xp[0], xb = xp[1];
            float xs[8] = {xa.x, xa.y, xa.z, xa.w, xb.x, xb.y, xb.z, xb.w};
            #pragma unroll
            for (int r = 0; r < 8; ++r) {
                float v = o[r] * xs[r] - o[r] + 1.0f;
                o[r] = v > 0.f ? v : 0.f;
            }
        }
        float4* dp = (float4*)(dst + outBase);
        dp[0] = make_float4(o[0], o[1], o[2], o[3]);
        dp[1] = make_float4(o[4], o[5], o[6], o[7]);
    }
}

extern "C" void kernel_launch(void* const* d_in, const int* in_sizes, int n_in,
                              void* d_out, int out_size, void* d_ws, size_t ws_size,
                              hipStream_t stream) {
    (void)in_sizes; (void)n_in; (void)out_size; (void)ws_size;
    const float* x  = (const float*)d_in[0];
    float* ws  = (float*)d_ws;
    float* out = (float*)d_out;
    const int grid = NB * (HH / ROWS) * (WW / TILE_X);

    // d_in layout: [0]=x, then w1,b1,w2,b2,...,w20,b20 (dict insertion order)
    auto W_ = [&](int i) { return (const float*)d_in[2 * i - 1]; };
    auto B_ = [&](int i) { return (const float*)d_in[2 * i];     };
    auto dstp = [&](int i)  { return ws + (size_t)(i  - 1) * LAYER_STRIDE; };
    auto srcp = [&](int L0) { return ws + (size_t)(L0 - 1) * LAYER_STRIDE; };

    // layer:                 (K, CIN, FINAL)  src(first concat layer)  dst
    conv_layer<1, 3,false><<<grid,128,0,stream>>>(x,       W_(1),  B_(1),  dstp(1),  nullptr);
    conv_layer<3, 3,false><<<grid,128,0,stream>>>(srcp(1), W_(2),  B_(2),  dstp(2),  nullptr);
    conv_layer<3, 6,false><<<grid,128,0,stream>>>(srcp(1), W_(3),  B_(3),  dstp(3),  nullptr);
    conv_layer<5, 6,false><<<grid,128,0,stream>>>(srcp(2), W_(4),  B_(4),  dstp(4),  nullptr);
    conv_layer<5, 6,false><<<grid,128,0,stream>>>(srcp(3), W_(5),  B_(5),  dstp(5),  nullptr);
    conv_layer<7, 9,false><<<grid,128,0,stream>>>(srcp(3), W_(6),  B_(6),  dstp(6),  nullptr);
    conv_layer<7,12,false><<<grid,128,0,stream>>>(srcp(3), W_(7),  B_(7),  dstp(7),  nullptr);
    conv_layer<5,12,false><<<grid,128,0,stream>>>(srcp(4), W_(8),  B_(8),  dstp(8),  nullptr);
    conv_layer<5,15,false><<<grid,128,0,stream>>>(srcp(4), W_(9),  B_(9),  dstp(9),  nullptr);
    conv_layer<3,15,false><<<grid,128,0,stream>>>(srcp(5), W_(10), B_(10), dstp(10), nullptr);
    conv_layer<3,18,false><<<grid,128,0,stream>>>(srcp(5), W_(11), B_(11), dstp(11), nullptr);
    conv_layer<3,18,false><<<grid,128,0,stream>>>(srcp(6), W_(12), B_(12), dstp(12), nullptr);
    conv_layer<3,21,false><<<grid,128,0,stream>>>(srcp(6), W_(13), B_(13), dstp(13), nullptr);
    conv_layer<3,21,false><<<grid,128,0,stream>>>(srcp(7), W_(14), B_(14), dstp(14), nullptr);
    conv_layer<3,24,false><<<grid,128,0,stream>>>(srcp(7), W_(15), B_(15), dstp(15), nullptr);
    conv_layer<3,24,false><<<grid,128,0,stream>>>(srcp(8), W_(16), B_(16), dstp(16), nullptr);
    conv_layer<3,24,false><<<grid,128,0,stream>>>(srcp(9), W_(17), B_(17), dstp(17), nullptr);
    conv_layer<3,27,false><<<grid,128,0,stream>>>(srcp(9), W_(18), B_(18), dstp(18), nullptr);
    conv_layer<3,30,false><<<grid,128,0,stream>>>(srcp(9), W_(19), B_(19), dstp(19), nullptr);
    conv_layer<3,57,true ><<<grid,128,0,stream>>>(srcp(1), W_(20), B_(20), out,      x);
}
